// AttentionBlock_8916352106770
// MI455X (gfx1250) — compile-verified, matrix operand order checked by reading, not run on hardware
//
#include <hip/hip_runtime.h>
#include <hip/hip_bf16.h>
#include <stdint.h>

typedef __bf16 bf16_t;
typedef __attribute__((ext_vector_type(16))) __bf16 v16bf;
typedef __attribute__((ext_vector_type(8)))  __bf16 v8bf;
typedef __attribute__((ext_vector_type(2)))  __bf16 v2bf;
typedef __attribute__((ext_vector_type(8)))  float  v8f;
typedef __attribute__((ext_vector_type(4)))  float  v4f;

#define B_SZ 4
#define SEQ  2048
#define DIM  256
#define NH   4
#define DH   64
#define ROWS (B_SZ * SEQ)   /* 8192 */

#define KLD 72   /* LDS leading dim (elements) for K tiles: 144B rows */

__device__ __forceinline__ void lds_fence() {
  // DS ops are in-order per wave (DScnt in-order); this blocks compiler
  // reordering and waits out the write before the dependent ds_load.
  asm volatile("s_wait_dscnt 0x0" ::: "memory");
}

// Async global->LDS copy, 16B per lane, tracked with ASYNCcnt.
__device__ __forceinline__ void async_ld_b128(uint32_t lds_byte_off, const void* g) {
  asm volatile("global_load_async_to_lds_b128 %0, %1, off"
               :: "v"(lds_byte_off), "v"(g) : "memory");
}
#define WAIT_ASYNC(n) asm volatile("s_wait_asynccnt " #n ::: "memory")

__device__ __forceinline__ v8f wmma_bf16(v16bf a, v16bf b, v8f c) {
  return __builtin_amdgcn_wmma_f32_16x16x32_bf16(false, a, false, b,
                                                 (short)0, c, false, false);
}

// A-operand (or B^T-operand) fragment, K=32, 16-bit elements.
// Element (r = lane&15, k) read from row-major memory [r][k], leading dim `ld`
// (elements). Lane's data: k = (lane>>4)*8 + {0..7} and +16..+23 — two
// contiguous 16B chunks, matching ISA 16-bit A layout (CDNA5 ISA 7.12.2).
__device__ __forceinline__ v16bf ld_frag_bf16(const bf16_t* base, int ld) {
  const int lane = threadIdx.x & 31;
  const bf16_t* p = base + (lane & 15) * ld + (lane >> 4) * 8;
  v8bf lo = *(const v8bf*)p;
  v8bf hi = *(const v8bf*)(p + 16);
  return __builtin_shufflevector(lo, hi, 0,1,2,3,4,5,6,7,8,9,10,11,12,13,14,15);
}

// Same pattern, but source is f32; convert to bf16 in-register.
__device__ __forceinline__ v16bf ld_frag_f32(const float* base, int ld) {
  const int lane = threadIdx.x & 31;
  const float* p = base + (lane & 15) * ld + (lane >> 4) * 8;
  v4f a0 = *(const v4f*)p;
  v4f a1 = *(const v4f*)(p + 4);
  v4f a2 = *(const v4f*)(p + 16);
  v4f a3 = *(const v4f*)(p + 20);
  v16bf f;
#pragma unroll
  for (int j = 0; j < 4; ++j) {
    f[j]      = (bf16_t)a0[j];
    f[4 + j]  = (bf16_t)a1[j];
    f[8 + j]  = (bf16_t)a2[j];
    f[12 + j] = (bf16_t)a3[j];
  }
  return f;
}

// ---------------------------------------------------------------------------
// Kernel 1: fused QKV projections.  z=0: Qp = Q@Wq^T+bq (f32 + bf16 outputs)
//           z=1: Kp = K@Wk^T+bk (bf16)   z=2: Vp = K@Wv^T+bv (bf16)
// One wave (32 threads) computes a 16x64 output tile (4 accumulators).
// ---------------------------------------------------------------------------
__global__ __launch_bounds__(32) void proj_kernel(
    const float* __restrict__ Qin, const float* __restrict__ Kin,
    const float* __restrict__ Wq, const float* __restrict__ bq,
    const float* __restrict__ Wk, const float* __restrict__ bk,
    const float* __restrict__ Wv, const float* __restrict__ bv,
    bf16_t* __restrict__ Qpb, float* __restrict__ Qpf,
    bf16_t* __restrict__ Kpb, bf16_t* __restrict__ Vpb)
{
  const int z = blockIdx.z;
  const float* X; const float* W; const float* bias;
  bf16_t* outb; float* outf = nullptr;
  if (z == 0)      { X = Qin; W = Wq; bias = bq; outb = Qpb; outf = Qpf; }
  else if (z == 1) { X = Kin; W = Wk; bias = bk; outb = Kpb; }
  else             { X = Kin; W = Wv; bias = bv; outb = Vpb; }

  const int row0 = blockIdx.x * 16;
  const int col0 = blockIdx.y * 64;
  v8f acc[4] = {};
  for (int k0 = 0; k0 < DIM; k0 += 32) {
    const v16bf a = ld_frag_f32(X + (size_t)row0 * DIM + k0, DIM);
#pragma unroll
    for (int t = 0; t < 4; ++t) {
      // B^T rows == W rows (out = X @ W^T), so B-frag = same row pattern on W.
      const v16bf b = ld_frag_f32(W + (size_t)(col0 + t * 16) * DIM + k0, DIM);
      acc[t] = wmma_bf16(a, b, acc[t]);
    }
  }
  const int lane = threadIdx.x & 31;
  const int hl = lane >> 4, cn = lane & 15;
#pragma unroll
  for (int t = 0; t < 4; ++t) {
    const int col = col0 + t * 16 + cn;
    const float bb = bias[col];
#pragma unroll
    for (int i = 0; i < 8; ++i) {             // C layout: row = i + 8*hl
      const int row = row0 + i + 8 * hl;
      const float v = acc[t][i] + bb;
      outb[(size_t)row * DIM + col] = (bf16_t)v;
      if (outf) outf[(size_t)row * DIM + col] = v;
    }
  }
}

// ---------------------------------------------------------------------------
// Kernel 2: flash attention, one wave per (b, h, 16-query tile), 4 waves/WG.
// K tiles are staged into LDS with double-buffered global_load_async_to_lds
// (ASYNCcnt); V tiles staged transposed via ds stores; P re-laid out via LDS.
// Per 32-key block: 4 WMMAs for S = Q K^T, online softmax, 4 WMMAs for P V.
// ---------------------------------------------------------------------------
__global__ __launch_bounds__(128) void attn_kernel(
    const bf16_t* __restrict__ Qp, const bf16_t* __restrict__ Kp,
    const bf16_t* __restrict__ Vp, const float* __restrict__ Qf,
    float* __restrict__ Octx)
{
  // Row strides padded so every ds_load_b128 row is 16B-aligned and the 16
  // rows a fragment touches land in distinct banks.
  __shared__ __attribute__((aligned(16))) bf16_t sK [4][2][32 * KLD]; // [key][dh]
  __shared__ __attribute__((aligned(16))) bf16_t sVt[4][DH * 40];     // [dh][key]
  __shared__ __attribute__((aligned(16))) bf16_t sP [4][16 * 40];     // [row][key]
  const int wave = threadIdx.x >> 5;
  const int lane = threadIdx.x & 31;
  const int tile = blockIdx.x * 4 + wave;     // 0..2047
  const int bh = tile >> 7;                   // 128 q-tiles per (b,h)
  const int b = bh >> 2, h = bh & 3;
  const int q0 = (tile & 127) * 16;
  const int hl = lane >> 4, cn = lane & 15;

  const bf16_t* qbase = Qp + ((size_t)(b * SEQ + q0)) * DIM + h * DH;
  const v16bf aq0 = ld_frag_bf16(qbase, DIM);        // dh 0..31
  const v16bf aq1 = ld_frag_bf16(qbase + 32, DIM);   // dh 32..63

  v8f acc[4] = {};
  float m_r[8], l_r[8];
#pragma unroll
  for (int i = 0; i < 8; ++i) { m_r[i] = -3.0e30f; l_r[i] = 0.f; }

  bf16_t* vbuf = sVt[wave];
  bf16_t* pbuf = sP[wave];
  const uint32_t kbase_lds[2] = {
      (uint32_t)(uintptr_t)&sK[wave][0][0],
      (uint32_t)(uintptr_t)&sK[wave][1][0] };

  // Async-stage one 32x64 bf16 K tile: 8 x b128 per wave (4 rows per issue).
  const int akey = lane >> 3;                 // 0..3
  const int aseg = (lane & 7) * 16;           // byte offset within 128B row
  const char* kptr = (const char*)(Kp + ((size_t)(b * SEQ)) * DIM + h * DH);
#define STAGE_K(kb_, buf_)                                                   \
  {                                                                          \
    _Pragma("unroll")                                                        \
    for (int kk = 0; kk < 32; kk += 4) {                                     \
      const int key = kk + akey;                                             \
      async_ld_b128(kbase_lds[buf_] + (uint32_t)(key * (KLD * 2) + aseg),    \
                    kptr + ((size_t)((kb_) + key)) * (DIM * 2) + aseg);      \
    }                                                                        \
  }

  STAGE_K(0, 0);
  int p = 0;

  for (int kb = 0; kb < SEQ; kb += 32) {
    const int have_next = (kb + 32 < SEQ);
    if (have_next) {
      if (p == 0) STAGE_K(kb + 32, 1) else STAGE_K(kb + 32, 0);
    }
    // Stage V tile transposed while async K copy is in flight.
#pragma unroll 4
    for (int kk = 0; kk < 32; ++kk) {
      const uint32_t* src =
          (const uint32_t*)(Vp + ((size_t)(b * SEQ + kb + kk)) * DIM + h * DH);
      v2bf pr = __builtin_bit_cast(v2bf, src[lane]);
      vbuf[(2 * lane) * 40 + kk]     = pr.x;
      vbuf[(2 * lane + 1) * 40 + kk] = pr.y;
    }
    // Current K tile resident when <=8 async remain (the next tile's).
    if (have_next) { WAIT_ASYNC(0x8); } else { WAIT_ASYNC(0x0); }

    const bf16_t* kb0 = sK[wave][p];
    v8f s0 = {}, s1 = {};
    s0 = wmma_bf16(aq0, ld_frag_bf16(kb0, KLD), s0);
    s0 = wmma_bf16(aq1, ld_frag_bf16(kb0 + 32, KLD), s0);
    s1 = wmma_bf16(aq0, ld_frag_bf16(kb0 + 16 * KLD, KLD), s1);
    s1 = wmma_bf16(aq1, ld_frag_bf16(kb0 + 16 * KLD + 32, KLD), s1);

    // Online softmax; row of C tile = VGPR i (+8 for upper half-wave),
    // N striped over 16 lanes -> xor-reduce within 16-lane halves.
#pragma unroll
    for (int i = 0; i < 8; ++i) {
      const float x0 = s0[i] * 0.125f, x1 = s1[i] * 0.125f; // 1/sqrt(64)
      float mx = fmaxf(x0, x1);
      mx = fmaxf(mx, __shfl_xor(mx, 1, 32));
      mx = fmaxf(mx, __shfl_xor(mx, 2, 32));
      mx = fmaxf(mx, __shfl_xor(mx, 4, 32));
      mx = fmaxf(mx, __shfl_xor(mx, 8, 32));
      const float mnew  = fmaxf(m_r[i], mx);
      const float alpha = __expf(m_r[i] - mnew);
      const float p0 = __expf(x0 - mnew);
      const float p1 = __expf(x1 - mnew);
      float rs = p0 + p1;
      rs += __shfl_xor(rs, 1, 32);
      rs += __shfl_xor(rs, 2, 32);
      rs += __shfl_xor(rs, 4, 32);
      rs += __shfl_xor(rs, 8, 32);
      l_r[i] = l_r[i] * alpha + rs;
      m_r[i] = mnew;
#pragma unroll
      for (int t = 0; t < 4; ++t) acc[t][i] *= alpha;
      // C-layout -> A-layout via LDS: P[row][key]
      pbuf[(i + 8 * hl) * 40 + cn]      = (bf16_t)p0;
      pbuf[(i + 8 * hl) * 40 + 16 + cn] = (bf16_t)p1;
    }
    lds_fence();
    const v16bf ap = ld_frag_bf16(pbuf, 40);          // A = P (16x32 keys)
#pragma unroll
    for (int t = 0; t < 4; ++t) {                     // B^T = V^T rows (dh)
      const v16bf bv = ld_frag_bf16(vbuf + t * 16 * 40, 40);
      acc[t] = wmma_bf16(ap, bv, acc[t]);
    }
    p ^= 1;
  }
#undef STAGE_K

  float inv[8];
#pragma unroll
  for (int i = 0; i < 8; ++i) inv[i] = 1.0f / l_r[i];
#pragma unroll
  for (int t = 0; t < 4; ++t)
#pragma unroll
    for (int i = 0; i < 8; ++i) {
      const int r = q0 + i + 8 * hl;
      const int c = h * DH + t * 16 + cn;
      const size_t idx = ((size_t)(b * SEQ + r)) * DIM + c;
      Octx[idx] = Qf[idx] + acc[t][i] * inv[i];       // residual: Qp + ctx
    }
}

// ---------------------------------------------------------------------------
// Kernel 3: O = LN(Octx); O = O + relu(O@Wo^T + bo); O = LN(O).
// 256 threads handle 16 rows; Wo GEMM: 8 waves x (16x32) WMMA tiles.
// ---------------------------------------------------------------------------
__global__ __launch_bounds__(256) void post_kernel(
    const float* __restrict__ Octx, const float* __restrict__ Wo,
    const float* __restrict__ bo,
    const float* __restrict__ g0, const float* __restrict__ be0,
    const float* __restrict__ g1, const float* __restrict__ be1,
    float* __restrict__ Out)
{
  __shared__ __attribute__((aligned(16))) float  sO1 [16 * DIM];
  __shared__ __attribute__((aligned(16))) bf16_t sO1b[16 * 264];
  __shared__ __attribute__((aligned(16))) float  sO2 [16 * DIM];
  __shared__ float red0[16][16];
  __shared__ float red1[16][16];
  __shared__ float stats[16][2];
  const int t = threadIdx.x;
  const int row = t >> 4;
  const int sub = t & 15;
  const int grow0 = blockIdx.x * 16;

  // ---- LayerNorm #1 ----
  float x[16];
  {
    const float* src = Octx + ((size_t)(grow0 + row)) * DIM + sub * 16;
    float s = 0.f, ss = 0.f;
#pragma unroll
    for (int j = 0; j < 16; ++j) { x[j] = src[j]; s += x[j]; ss += x[j] * x[j]; }
    red0[row][sub] = s; red1[row][sub] = ss;
  }
  __syncthreads();
  if (sub == 0) {
    float S = 0.f, SS = 0.f;
#pragma unroll
    for (int j = 0; j < 16; ++j) { S += red0[row][j]; SS += red1[row][j]; }
    const float m = S * (1.f / DIM);
    stats[row][0] = m;
    stats[row][1] = rsqrtf(SS * (1.f / DIM) - m * m + 1e-5f);
  }
  __syncthreads();
  {
    const float mean = stats[row][0], rstd = stats[row][1];
#pragma unroll
    for (int j = 0; j < 16; ++j) {
      const int col = sub * 16 + j;
      const float v = (x[j] - mean) * rstd * g0[col] + be0[col];
      sO1 [row * DIM + col] = v;
      sO1b[row * 264 + col] = (bf16_t)v;
    }
  }
  __syncthreads();

  // ---- R = relu(O1 @ Wo^T + bo); O2 = O1 + R ----
  const int wave = t >> 5, lane = t & 31;
  const int hl = lane >> 4, cn = lane & 15;
  v8f acc[2] = {};
  for (int k0 = 0; k0 < DIM; k0 += 32) {
    const v16bf a = ld_frag_bf16(sO1b + k0, 264);
#pragma unroll
    for (int u = 0; u < 2; ++u) {
      const v16bf bw = ld_frag_f32(Wo + (size_t)(wave * 32 + u * 16) * DIM + k0, DIM);
      acc[u] = wmma_bf16(a, bw, acc[u]);
    }
  }
#pragma unroll
  for (int u = 0; u < 2; ++u) {
    const int col = wave * 32 + u * 16 + cn;
    const float bb = bo[col];
#pragma unroll
    for (int i = 0; i < 8; ++i) {
      const int r = i + 8 * hl;
      float rv = acc[u][i] + bb;
      rv = rv > 0.f ? rv : 0.f;
      sO2[r * DIM + col] = sO1[r * DIM + col] + rv;
    }
  }
  __syncthreads();

  // ---- LayerNorm #2, write output ----
  float y[16];
  {
    float s = 0.f, ss = 0.f;
#pragma unroll
    for (int j = 0; j < 16; ++j) {
      y[j] = sO2[row * DIM + sub * 16 + j];
      s += y[j]; ss += y[j] * y[j];
    }
    red0[row][sub] = s; red1[row][sub] = ss;
  }
  __syncthreads();
  if (sub == 0) {
    float S = 0.f, SS = 0.f;
#pragma unroll
    for (int j = 0; j < 16; ++j) { S += red0[row][j]; SS += red1[row][j]; }
    const float m = S * (1.f / DIM);
    stats[row][0] = m;
    stats[row][1] = rsqrtf(SS * (1.f / DIM) - m * m + 1e-5f);
  }
  __syncthreads();
  {
    const float mean = stats[row][0], rstd = stats[row][1];
    float* dst = Out + ((size_t)(grow0 + row)) * DIM + sub * 16;
#pragma unroll
    for (int j = 0; j < 16; ++j) {
      const int col = sub * 16 + j;
      dst[j] = (y[j] - mean) * rstd * g1[col] + be1[col];
    }
  }
}

extern "C" void kernel_launch(void* const* d_in, const int* in_sizes, int n_in,
                              void* d_out, int out_size, void* d_ws, size_t ws_size,
                              hipStream_t stream) {
  (void)in_sizes; (void)n_in; (void)out_size; (void)ws_size;
  const float* Qin = (const float*)d_in[0];
  const float* Kin = (const float*)d_in[1];
  const float* Wq  = (const float*)d_in[2];
  const float* bq  = (const float*)d_in[3];
  const float* Wk  = (const float*)d_in[4];
  const float* bk  = (const float*)d_in[5];
  const float* Wv  = (const float*)d_in[6];
  const float* bv  = (const float*)d_in[7];
  const float* Wo  = (const float*)d_in[8];
  const float* bo  = (const float*)d_in[9];
  const float* g0  = (const float*)d_in[10];
  const float* be0 = (const float*)d_in[11];
  const float* g1  = (const float*)d_in[12];
  const float* be1 = (const float*)d_in[13];
  float* out = (float*)d_out;

  // Workspace: Qp/Kp/Vp bf16 (4MB ea) + Qp f32 (8MB) + Octx f32 (8MB) = 28MB
  char* ws = (char*)d_ws;
  size_t off = 0;
  bf16_t* Qpb = (bf16_t*)(ws + off); off += (size_t)ROWS * DIM * sizeof(bf16_t);
  bf16_t* Kpb = (bf16_t*)(ws + off); off += (size_t)ROWS * DIM * sizeof(bf16_t);
  bf16_t* Vpb = (bf16_t*)(ws + off); off += (size_t)ROWS * DIM * sizeof(bf16_t);
  float*  Qpf = (float*) (ws + off); off += (size_t)ROWS * DIM * sizeof(float);
  float*  Oc  = (float*) (ws + off); off += (size_t)ROWS * DIM * sizeof(float);

  proj_kernel<<<dim3(ROWS / 16, DIM / 64, 3), 32, 0, stream>>>(
      Qin, Kin, Wq, bq, Wk, bk, Wv, bv, Qpb, Qpf, Kpb, Vpb);
  attn_kernel<<<dim3((B_SZ * NH * (SEQ / 16)) / 4), 128, 0, stream>>>(
      Qpb, Kpb, Vpb, Qpf, Oc);
  post_kernel<<<dim3(ROWS / 16), 256, 0, stream>>>(
      Oc, Wo, bo, g0, be0, g1, be1, out);
}
